// LayerNormLSTM_46325517254840
// MI455X (gfx1250) — compile-verified, matrix operand order checked by reading, not run on hardware
//
#include <hip/hip_runtime.h>

typedef __attribute__((ext_vector_type(16))) __bf16 v16bf;
typedef __attribute__((ext_vector_type(8)))  float  v8f;

#define TT 256   // timesteps
#define DD 512   // input dim
#define HH 512   // hidden
#define GG 2048  // 4H
#define EPS 1e-5f

// ---------- helpers ----------
__device__ __forceinline__ unsigned short f2bf(float f) {
  unsigned u = __float_as_uint(f);
  u += 0x7FFFu + ((u >> 16) & 1u);           // round-to-nearest-even
  return (unsigned short)(u >> 16);
}
__device__ __forceinline__ float sigm(float x) { return 1.0f / (1.0f + __expf(-x)); }

// ---------- weight pre-pack: fp32 [4H,512] -> bf16 WMMA B-fragment layout ----------
// dst layout: [mat(2)][tileN(128)][kc(16)][lane(32)][v(16)]
// value v of lane l maps to: n = tileN*16 + (l&15),
//                            k = kc*32 + ((v>>3)*16) + (v&7) + ((l>>4)*8)
__global__ void pack_weights(const float* __restrict__ Wih,
                             const float* __restrict__ Whh,
                             unsigned short* __restrict__ dst) {
  int idx = blockIdx.x * blockDim.x + threadIdx.x;
  if (idx >= 2 * GG * DD) return;
  int mat = idx >> 20;                 // 2048*512 = 1<<20
  int r   = idx & ((1 << 20) - 1);
  int v    = r & 15;  r >>= 4;
  int lane = r & 31;  r >>= 5;
  int kc   = r & 15;  r >>= 4;
  int tileN = r;                       // 0..127
  int n = tileN * 16 + (lane & 15);
  int k = kc * 32 + ((v >> 3) * 16) + (v & 7) + ((lane >> 4) * 8);
  const float* W = mat ? Whh : Wih;
  dst[idx] = f2bf(W[(size_t)n * DD + k]);
}

// ---------- zero h (bf16) and c (fp32) ----------
__global__ void init_state(unsigned int* __restrict__ hbf_dw, float* __restrict__ c) {
  int i = blockIdx.x * blockDim.x + threadIdx.x;   // 131072 threads
  if (i < 65536) hbf_dw[i] = 0u;                   // 131072 bf16 = 65536 dwords
  if (i < 131072) c[i] = 0.0f;
}

// ---------- fused LN-LSTM step: 16 WGs x 512 threads, 16 rows per WG ----------
__global__ __launch_bounds__(512, 1)
void lnlstm_step(const float* __restrict__ x,
                 const float* __restrict__ b_ih, const float* __restrict__ b_hh,
                 const float* __restrict__ g_ih, const float* __restrict__ be_ih,
                 const float* __restrict__ g_hh, const float* __restrict__ be_hh,
                 const float* __restrict__ g_c,  const float* __restrict__ be_c,
                 const unsigned short* __restrict__ Wih_p,
                 const unsigned short* __restrict__ Whh_p,
                 unsigned short* __restrict__ h_bf,
                 float* __restrict__ c_ws,
                 float* __restrict__ out, int t) {
  __shared__ unsigned int sX[4096];   // 16 rows x 512 bf16 (pairs)
  __shared__ unsigned int sH[4096];
  __shared__ float rsum_i[16], rq_i[16], rsum_h[16], rq_h[16], csum[16], cq[16];

  const int tid  = threadIdx.x;
  const int w    = tid >> 5;        // wave 0..15
  const int lane = tid & 31;
  const int half = lane >> 4;
  const int ln   = lane & 15;
  const int row0 = blockIdx.x * 16;

  if (tid < 16) {
    rsum_i[tid] = 0.f; rq_i[tid] = 0.f;
    rsum_h[tid] = 0.f; rq_h[tid] = 0.f;
    csum[tid]   = 0.f; cq[tid]   = 0.f;
  }

  // stage x_t (fp32 -> bf16) and h (bf16 copy) into LDS, row-major [16][512]
  const unsigned int* hsrc = (const unsigned int*)h_bf;
  for (int i = tid; i < 4096; i += 512) {
    int r  = i >> 8;            // row in WG tile
    int dw = i & 255;           // dword within row (2 elems)
    const float* xp = x + (((size_t)(row0 + r) * TT + t) * DD + dw * 2);
    sX[i] = (unsigned)f2bf(xp[0]) | ((unsigned)f2bf(xp[1]) << 16);
    sH[i] = hsrc[(size_t)(row0 + r) * 256 + dw];
  }
  __syncthreads();

  // ---- dual GEMM over K=512 (16 chunks of 32), bf16 WMMA, fp32 accum ----
  v8f acc_i[8], acc_h[8];
  v8f zero = {0.f, 0.f, 0.f, 0.f, 0.f, 0.f, 0.f, 0.f};
#pragma unroll
  for (int a = 0; a < 8; ++a) { acc_i[a] = zero; acc_h[a] = zero; }

  for (int kc = 0; kc < 16; ++kc) {
    union { v16bf v; uint4 u[2]; } ax, ah;
    int b0 = ln * 512 + kc * 32 + half * 8;          // bf16 index, 16B aligned
    ax.u[0] = *(const uint4*)&sX[b0 >> 1];
    ax.u[1] = *(const uint4*)&sX[(b0 + 16) >> 1];
    ah.u[0] = *(const uint4*)&sH[b0 >> 1];
    ah.u[1] = *(const uint4*)&sH[(b0 + 16) >> 1];
#pragma unroll
    for (int q = 0; q < 4; ++q) {
#pragma unroll
      for (int s = 0; s < 2; ++s) {
        int tileN = q * 32 + w * 2 + s;
        size_t woff = (((size_t)tileN * 16 + kc) * 32 + lane) * 16;   // bf16 elems
        union { v16bf v; uint4 u[2]; } bi, bh;
        bi.u[0] = *(const uint4*)(Wih_p + woff);
        bi.u[1] = *(const uint4*)(Wih_p + woff + 8);
        bh.u[0] = *(const uint4*)(Whh_p + woff);
        bh.u[1] = *(const uint4*)(Whh_p + woff + 8);
        int a = q * 2 + s;
        acc_i[a] = __builtin_amdgcn_wmma_f32_16x16x32_bf16(
            false, ax.v, false, bi.v, (short)0, acc_i[a], false, false);
        acc_h[a] = __builtin_amdgcn_wmma_f32_16x16x32_bf16(
            false, ah.v, false, bh.v, (short)0, acc_h[a], false, false);
      }
    }
  }

  // ---- add biases (part of LN input) ----
#pragma unroll
  for (int a = 0; a < 8; ++a) {
    int tileN = (a >> 1) * 32 + w * 2 + (a & 1);
    int col = tileN * 16 + ln;
    float bi = b_ih[col], bh = b_hh[col];
#pragma unroll
    for (int v = 0; v < 8; ++v) { acc_i[a][v] += bi; acc_h[a][v] += bh; }
  }

  // ---- per-row mean/var over 2048 gate columns (wave-local partials) ----
  float s_i[8], q_i[8], s_h[8], q_h[8];
#pragma unroll
  for (int v = 0; v < 8; ++v) { s_i[v] = 0.f; q_i[v] = 0.f; s_h[v] = 0.f; q_h[v] = 0.f; }
#pragma unroll
  for (int a = 0; a < 8; ++a) {
#pragma unroll
    for (int v = 0; v < 8; ++v) {
      float ai = acc_i[a][v], ah2 = acc_h[a][v];
      s_i[v] += ai; q_i[v] += ai * ai;
      s_h[v] += ah2; q_h[v] += ah2 * ah2;
    }
  }
#pragma unroll
  for (int m = 1; m < 16; m <<= 1) {
#pragma unroll
    for (int v = 0; v < 8; ++v) {
      s_i[v] += __shfl_xor(s_i[v], m, 32);
      q_i[v] += __shfl_xor(q_i[v], m, 32);
      s_h[v] += __shfl_xor(s_h[v], m, 32);
      q_h[v] += __shfl_xor(q_h[v], m, 32);
    }
  }
  if (ln == 0) {
#pragma unroll
    for (int v = 0; v < 8; ++v) {
      int r = half * 8 + v;
      atomicAdd(&rsum_i[r], s_i[v]); atomicAdd(&rq_i[r], q_i[v]);
      atomicAdd(&rsum_h[r], s_h[v]); atomicAdd(&rq_h[r], q_h[v]);
    }
  }
  __syncthreads();

  float mi[8], ri[8], mh[8], rh[8];
#pragma unroll
  for (int v = 0; v < 8; ++v) {
    int r = half * 8 + v;
    float m  = rsum_i[r] * (1.0f / GG);
    float vv = rq_i[r] * (1.0f / GG) - m * m;
    mi[v] = m; ri[v] = rsqrtf(vv + EPS);
    m  = rsum_h[r] * (1.0f / GG);
    vv = rq_h[r] * (1.0f / GG) - m * m;
    mh[v] = m; rh[v] = rsqrtf(vv + EPS);
  }

  // ---- LN + gate combine + cell update (all wave-local in registers) ----
  float cvals[2][8], ovals[2][8];
#pragma unroll
  for (int s = 0; s < 2; ++s) {
    int j = 32 * w + 16 * s + ln;       // hidden index 0..511
#pragma unroll
    for (int v = 0; v < 8; ++v) {
      float gt[4];
#pragma unroll
      for (int q = 0; q < 4; ++q) {
        int col = (q * 32 + w * 2 + s) * 16 + ln;
        float gi = (acc_i[q * 2 + s][v] - mi[v]) * ri[v] * g_ih[col] + be_ih[col];
        float gh = (acc_h[q * 2 + s][v] - mh[v]) * rh[v] * g_hh[col] + be_hh[col];
        gt[q] = gi + gh;
      }
      int grow = row0 + half * 8 + v;
      float cold = c_ws[(size_t)grow * HH + j];
      float cn = sigm(gt[1]) * cold + sigm(gt[0]) * tanhf(gt[2]);
      c_ws[(size_t)grow * HH + j] = cn;
      cvals[s][v] = cn;
      ovals[s][v] = sigm(gt[3]);
    }
  }

  // ---- c LayerNorm over H=512 ----
  float cs[8], cq2[8];
#pragma unroll
  for (int v = 0; v < 8; ++v) {
    cs[v]  = cvals[0][v] + cvals[1][v];
    cq2[v] = cvals[0][v] * cvals[0][v] + cvals[1][v] * cvals[1][v];
  }
#pragma unroll
  for (int m = 1; m < 16; m <<= 1) {
#pragma unroll
    for (int v = 0; v < 8; ++v) {
      cs[v]  += __shfl_xor(cs[v], m, 32);
      cq2[v] += __shfl_xor(cq2[v], m, 32);
    }
  }
  if (ln == 0) {
#pragma unroll
    for (int v = 0; v < 8; ++v) {
      int r = half * 8 + v;
      atomicAdd(&csum[r], cs[v]); atomicAdd(&cq[r], cq2[v]);
    }
  }
  __syncthreads();

#pragma unroll
  for (int v = 0; v < 8; ++v) {
    int r = half * 8 + v;
    float mc = csum[r] * (1.0f / HH);
    float vc = cq[r] * (1.0f / HH) - mc * mc;
    float rc = rsqrtf(vc + EPS);
    int grow = row0 + r;
#pragma unroll
    for (int s = 0; s < 2; ++s) {
      int j = 32 * w + 16 * s + ln;
      float cn = cvals[s][v];
      float hn = ovals[s][v] * tanhf((cn - mc) * rc * g_c[j] + be_c[j]);
      out[((size_t)grow * TT + t) * HH + j] = hn;           // outs[b][t][h]
      h_bf[(size_t)grow * HH + j] = f2bf(hn);               // next-step A operand
      if (t == TT - 1) {
        size_t tail = (size_t)256 * TT * HH;                // 33,554,432
        out[tail + (size_t)grow * HH + j] = hn;             // final h
        out[tail + (size_t)256 * HH + (size_t)grow * HH + j] = cn;  // final c
      }
    }
  }
}

extern "C" void kernel_launch(void* const* d_in, const int* in_sizes, int n_in,
                              void* d_out, int out_size, void* d_ws, size_t ws_size,
                              hipStream_t stream) {
  const float* x     = (const float*)d_in[0];
  const float* W_ih  = (const float*)d_in[1];
  const float* W_hh  = (const float*)d_in[2];
  const float* b_ih  = (const float*)d_in[3];
  const float* b_hh  = (const float*)d_in[4];
  const float* g_ih  = (const float*)d_in[5];
  const float* be_ih = (const float*)d_in[6];
  const float* g_hh  = (const float*)d_in[7];
  const float* be_hh = (const float*)d_in[8];
  const float* g_c   = (const float*)d_in[9];
  const float* be_c  = (const float*)d_in[10];

  char* ws = (char*)d_ws;
  unsigned short* wpack = (unsigned short*)ws;                         // 2 x 2 MB bf16
  unsigned short* hbf   = (unsigned short*)(ws + 4u * 1024 * 1024);    // 256 KB
  float*          c_ws  = (float*)(ws + 4u * 1024 * 1024 + 256u * 1024); // 512 KB
  float* out = (float*)d_out;

  pack_weights<<<8192, 256, 0, stream>>>(W_ih, W_hh, wpack);
  init_state<<<512, 256, 0, stream>>>((unsigned int*)hbf, c_ws);

  for (int t = 0; t < TT; ++t) {
    lnlstm_step<<<16, 512, 0, stream>>>(x, b_ih, b_hh, g_ih, be_ih, g_hh, be_hh,
                                        g_c, be_c, wpack, wpack + (1u << 20),
                                        hbf, c_ws, out, t);
  }
}
